// LSTMDecoder_41824391529094
// MI455X (gfx1250) — matrix operand order repaired and checked by reading, not compiled, not run on hardware
//
#include <hip/hip_runtime.h>
#include <hip/hip_bf16.h>

// ---------------- types for WMMA ----------------
typedef __attribute__((ext_vector_type(8)))  __bf16 v8bf;
typedef __attribute__((ext_vector_type(16))) __bf16 v16bf;
typedef __attribute__((ext_vector_type(8)))  float  v8f;

// ---------------- helpers ----------------
__device__ __forceinline__ unsigned short f2bf(float f) {
    unsigned int u = __float_as_uint(f);
    unsigned int r = (u + 0x7FFFu + ((u >> 16) & 1u)) >> 16;
    return (unsigned short)r;
}
__device__ __forceinline__ float bf2f(unsigned short h) {
    return __uint_as_float(((unsigned int)h) << 16);
}
__device__ __forceinline__ float sigm(float x) { return 1.0f / (1.0f + expf(-x)); }

#define NROWS 32768
#define DIM   1024

// ---------------- fp32 -> bf16 convert ----------------
__global__ void k_cvt_bf16(const float* __restrict__ in, unsigned short* __restrict__ out, int n) {
    int i = blockIdx.x * blockDim.x + threadIdx.x;
    int stride = gridDim.x * blockDim.x;
    for (; i < n; i += stride) out[i] = f2bf(in[i]);
}

// ---------------- LSTM single step, one layer ----------------
// grid = 1024 (hidden index k), block = 256 (4 gates x 64 lanes)
__global__ void __launch_bounds__(256)
k_lstm(const float* __restrict__ Wih, const float* __restrict__ Whh,
       const float* __restrict__ bih, const float* __restrict__ bhh,
       const float* __restrict__ x,   const float* __restrict__ h0,
       const float* __restrict__ c0,
       float* __restrict__ h_out, float* __restrict__ c_out) {
    int k = blockIdx.x;
    int tid = threadIdx.x;
    int g = tid >> 6;          // gate 0..3 (i,f,g,o)
    int q = tid & 63;          // lane within gate
    const float* wrx = Wih + (size_t)(g * DIM + k) * DIM + q * 16;
    const float* wrh = Whh + (size_t)(g * DIM + k) * DIM + q * 16;
    const float* xv  = x  + q * 16;
    const float* hv  = h0 + q * 16;
    float s = 0.0f;
#pragma unroll
    for (int i = 0; i < 16; i += 4) {
        float4 w0 = *(const float4*)(wrx + i);
        float4 a0 = *(const float4*)(xv + i);
        s += w0.x * a0.x + w0.y * a0.y + w0.z * a0.z + w0.w * a0.w;
        float4 w1 = *(const float4*)(wrh + i);
        float4 a1 = *(const float4*)(hv + i);
        s += w1.x * a1.x + w1.y * a1.y + w1.z * a1.z + w1.w * a1.w;
    }
    __shared__ float red[256];
    red[tid] = s;
    __syncthreads();
    for (int off = 32; off >= 1; off >>= 1) {
        if (q < off) red[tid] += red[tid + off];
        __syncthreads();
    }
    if (tid == 0) {
        float gi = sigm(red[0]   + bih[k]            + bhh[k]);
        float gf = sigm(red[64]  + bih[DIM + k]      + bhh[DIM + k]);
        float gg = tanhf(red[128] + bih[2 * DIM + k] + bhh[2 * DIM + k]);
        float go = sigm(red[192] + bih[3 * DIM + k]  + bhh[3 * DIM + k]);
        float c = gf * c0[k] + gi * gg;
        float h = go * tanhf(c);
        c_out[k] = c;
        h_out[k] = h;
    }
}

// ---------------- small matvec: out[j] = W[j]·x + b1[j] + b2[j] + b3[j] ----------------
// grid = 1024 rows, block = 128
__global__ void __launch_bounds__(128)
k_matvec(const float* __restrict__ W, const float* __restrict__ x,
         const float* __restrict__ b1, const float* __restrict__ b2,
         const float* __restrict__ b3, float* __restrict__ out) {
    int j = blockIdx.x;
    int t = threadIdx.x;
    const float* wr = W + (size_t)j * DIM + t * 8;
    const float* xr = x + t * 8;
    float s = 0.0f;
#pragma unroll
    for (int i = 0; i < 8; i += 4) {
        float4 w = *(const float4*)(wr + i);
        float4 a = *(const float4*)(xr + i);
        s += w.x * a.x + w.y * a.y + w.z * a.z + w.w * a.w;
    }
    __shared__ float red[128];
    red[t] = s;
    __syncthreads();
    for (int off = 64; off >= 1; off >>= 1) {
        if (t < off) red[t] += red[t + off];
        __syncthreads();
    }
    if (t == 0) out[j] = red[0] + b1[j] + b2[j] + b3[j];
}

// ---------------- fragment loaders (contiguous k = 16h + s, same order as B) ----------------
__device__ __forceinline__ v16bf load_a_frag(const unsigned short* __restrict__ arow,
                                             int k0, int kb) {
    v8bf alo = *(const v8bf*)(arow + k0 + kb);
    v8bf ahi = *(const v8bf*)(arow + k0 + kb + 8);
    return __builtin_shufflevector(alo, ahi, 0, 1, 2, 3, 4, 5, 6, 7,
                                   8, 9, 10, 11, 12, 13, 14, 15);
}

// ---------------- bf16 WMMA GEMM: C16[M,1024] = A16[M,1024] x B16[1024,1024]^T ----------------
// block = 256 thr (8 waves stacked in M), wave tile 16x64, grid = (M/128, 1024/64).
// Software-pipelined: iteration i issues loads for i+1 before consuming i's fragments,
// so s_wait_loadcnt covers a 4-WMMA math group instead of stalling per load.
__global__ void __launch_bounds__(256)
k_gemm_bf16(const unsigned short* __restrict__ A16,
            const unsigned short* __restrict__ B16,
            unsigned short* __restrict__ C16) {
    const int K = DIM;
    int lane = threadIdx.x & 31;
    int wave = threadIdx.x >> 5;
    int mbase = blockIdx.x * 128 + wave * 16;
    int nbase = blockIdx.y * 64;
    int lm = lane & 15;
    int hi = lane >> 4;          // 0 or 1
    int kb = hi * 16;            // A-fragment K sub-chunk base
    const unsigned short* arow = A16 + (size_t)(mbase + lm) * K;
    const unsigned short* brow0 = B16 + (size_t)(nbase + 0 * 16 + lm) * K + hi * 16;
    const unsigned short* brow1 = B16 + (size_t)(nbase + 1 * 16 + lm) * K + hi * 16;
    const unsigned short* brow2 = B16 + (size_t)(nbase + 2 * 16 + lm) * K + hi * 16;
    const unsigned short* brow3 = B16 + (size_t)(nbase + 3 * 16 + lm) * K + hi * 16;

    v8f acc0 = {}, acc1 = {}, acc2 = {}, acc3 = {};

    // prologue: fragments for k0 = 0
    v16bf a  = load_a_frag(arow, 0, kb);
    v16bf b0 = *(const v16bf*)(brow0);
    v16bf b1 = *(const v16bf*)(brow1);
    v16bf b2 = *(const v16bf*)(brow2);
    v16bf b3 = *(const v16bf*)(brow3);

    for (int k0 = 0; k0 < K - 32; k0 += 32) {
        // issue next iteration's loads first
        int kn = k0 + 32;
        v16bf an  = load_a_frag(arow, kn, kb);
        v16bf bn0 = *(const v16bf*)(brow0 + kn);
        v16bf bn1 = *(const v16bf*)(brow1 + kn);
        v16bf bn2 = *(const v16bf*)(brow2 + kn);
        v16bf bn3 = *(const v16bf*)(brow3 + kn);
        // consume current fragments
        acc0 = __builtin_amdgcn_wmma_f32_16x16x32_bf16(false, a, false, b0, (short)0, acc0, false, false);
        acc1 = __builtin_amdgcn_wmma_f32_16x16x32_bf16(false, a, false, b1, (short)0, acc1, false, false);
        acc2 = __builtin_amdgcn_wmma_f32_16x16x32_bf16(false, a, false, b2, (short)0, acc2, false, false);
        acc3 = __builtin_amdgcn_wmma_f32_16x16x32_bf16(false, a, false, b3, (short)0, acc3, false, false);
        a = an; b0 = bn0; b1 = bn1; b2 = bn2; b3 = bn3;
    }
    // epilogue K-step
    acc0 = __builtin_amdgcn_wmma_f32_16x16x32_bf16(false, a, false, b0, (short)0, acc0, false, false);
    acc1 = __builtin_amdgcn_wmma_f32_16x16x32_bf16(false, a, false, b1, (short)0, acc1, false, false);
    acc2 = __builtin_amdgcn_wmma_f32_16x16x32_bf16(false, a, false, b2, (short)0, acc2, false, false);
    acc3 = __builtin_amdgcn_wmma_f32_16x16x32_bf16(false, a, false, b3, (short)0, acc3, false, false);

    int mrow = mbase + 8 * hi;
    v8f accs[4] = {acc0, acc1, acc2, acc3};
#pragma unroll
    for (int t = 0; t < 4; ++t) {
#pragma unroll
        for (int r = 0; r < 8; ++r) {
            C16[(size_t)(mrow + r) * DIM + nbase + t * 16 + lm] = f2bf(accs[t][r]);
        }
    }
}

// ---------------- fused row reduction: t[n] = [relu]( v · tanh(wh[n,:] + b + cov[n]*wc) + vb ) ----
// grid = 32768 rows, block = 256 (4 cols per thread)
__global__ void __launch_bounds__(256)
k_rowdot(const unsigned short* __restrict__ wh16, const float* __restrict__ b,
         const float* __restrict__ wc, const float* __restrict__ v,
         const float* __restrict__ vb, const float* __restrict__ cov,
         float* __restrict__ out, int do_relu) {
    int n = blockIdx.x;
    int t = threadIdx.x;
    int j = t * 4;
    float cn = cov[n];
    const unsigned short* row = wh16 + (size_t)n * DIM + j;
    unsigned long long w = *(const unsigned long long*)row;
    float4 bv = *(const float4*)(b + j);
    float4 cv = *(const float4*)(wc + j);
    float4 vv = *(const float4*)(v + j);
    float s = 0.0f;
    s += vv.x * tanhf(bf2f((unsigned short)(w)) + bv.x + cn * cv.x);
    s += vv.y * tanhf(bf2f((unsigned short)(w >> 16)) + bv.y + cn * cv.y);
    s += vv.z * tanhf(bf2f((unsigned short)(w >> 32)) + bv.z + cn * cv.z);
    s += vv.w * tanhf(bf2f((unsigned short)(w >> 48)) + bv.w + cn * cv.w);
    __shared__ float red[256];
    red[t] = s;
    __syncthreads();
    for (int off = 128; off >= 1; off >>= 1) {
        if (t < off) red[t] += red[t + off];
        __syncthreads();
    }
    if (t == 0) {
        float r = red[0] + vb[0];
        if (do_relu) r = fmaxf(r, 0.0f);
        out[n] = r;
    }
}

// ---------------- softmax reduce over 32768 values -> {max, sumexp} ----------------
__global__ void __launch_bounds__(1024)
k_smax_reduce(const float* __restrict__ t, float* __restrict__ mxsum) {
    int tid = threadIdx.x;
    __shared__ float red[1024];
    __shared__ float mx_s;
    float m = -1e30f;
    for (int i = tid; i < NROWS; i += 1024) m = fmaxf(m, t[i]);
    red[tid] = m;
    __syncthreads();
    for (int off = 512; off >= 1; off >>= 1) {
        if (tid < off) red[tid] = fmaxf(red[tid], red[tid + off]);
        __syncthreads();
    }
    if (tid == 0) mx_s = red[0];
    __syncthreads();
    float mx = mx_s;
    float s = 0.0f;
    for (int i = tid; i < NROWS; i += 1024) s += expf(t[i] - mx);
    red[tid] = s;
    __syncthreads();
    for (int off = 512; off >= 1; off >>= 1) {
        if (tid < off) red[tid] += red[tid + off];
        __syncthreads();
    }
    if (tid == 0) { mxsum[0] = mx; mxsum[1] = red[0]; }
}

// ---------------- softmax apply + coverage update ----------------
__global__ void k_smax_apply(const float* __restrict__ t, const float* __restrict__ mxsum,
                             const float* __restrict__ cov_in,
                             float* __restrict__ p_out, float* __restrict__ cov_out) {
    int n = blockIdx.x * blockDim.x + threadIdx.x;
    if (n < NROWS) {
        float p = expf(t[n] - mxsum[0]) / mxsum[1];
        p_out[n] = p;
        cov_out[n] = cov_in[n] + p;
    }
}

// ---------------- context init: ctx[j] = g_b1[j]  (since sum(attn)==1) ----------------
__global__ void k_ctx_init(const float* __restrict__ b1, float* __restrict__ ctx) {
    int j = blockIdx.x * blockDim.x + threadIdx.x;
    if (j < DIM) ctx[j] = b1[j];
}

// ---------------- context accumulate: ctx[j] += sum_n attn[n]*wh[n,j] ----------------
// grid = 128 blocks (256 rows each), block = 256 (4 cols per thread)
__global__ void __launch_bounds__(256)
k_ctx_accum(const unsigned short* __restrict__ wh16, const float* __restrict__ attn,
            float* __restrict__ ctx) {
    int t = threadIdx.x;
    int j0 = t * 4;
    float a0 = 0.f, a1 = 0.f, a2 = 0.f, a3 = 0.f;
    int nstart = blockIdx.x * 256;
    for (int n = nstart; n < nstart + 256; ++n) {
        float an = attn[n];
        unsigned long long w = *(const unsigned long long*)(wh16 + (size_t)n * DIM + j0);
        a0 += an * bf2f((unsigned short)(w));
        a1 += an * bf2f((unsigned short)(w >> 16));
        a2 += an * bf2f((unsigned short)(w >> 32));
        a3 += an * bf2f((unsigned short)(w >> 48));
    }
    atomicAdd(&ctx[j0 + 0], a0);
    atomicAdd(&ctx[j0 + 1], a1);
    atomicAdd(&ctx[j0 + 2], a2);
    atomicAdd(&ctx[j0 + 3], a3);
}

// =====================================================================
extern "C" void kernel_launch(void* const* d_in, const int* in_sizes, int n_in,
                              void* d_out, int out_size, void* d_ws, size_t ws_size,
                              hipStream_t stream) {
    (void)in_sizes; (void)n_in; (void)out_size; (void)ws_size;
    const float* in_x   = (const float*)d_in[0];
    const float* h0     = (const float*)d_in[1];
    const float* c0     = (const float*)d_in[2];
    const float* enc    = (const float*)d_in[3];
    const float* cov_g  = (const float*)d_in[4];
    const float* cov_p  = (const float*)d_in[5];
    const float* W_ih   = (const float*)d_in[6];
    const float* W_hh   = (const float*)d_in[7];
    const float* b_ih   = (const float*)d_in[8];
    const float* b_hh   = (const float*)d_in[9];
    const float* g_w1   = (const float*)d_in[10];
    const float* g_b1   = (const float*)d_in[11];
    const float* g_w2   = (const float*)d_in[12];
    const float* g_b2   = (const float*)d_in[13];
    const float* g_wc   = (const float*)d_in[14];
    const float* g_bc   = (const float*)d_in[15];
    const float* g_v    = (const float*)d_in[16];
    const float* g_vb   = (const float*)d_in[17];
    const float* p_w1   = (const float*)d_in[18];
    const float* p_b1   = (const float*)d_in[19];
    const float* p_w2   = (const float*)d_in[20];
    const float* p_b2   = (const float*)d_in[21];
    const float* p_wc   = (const float*)d_in[22];
    const float* p_bc   = (const float*)d_in[23];
    const float* p_v    = (const float*)d_in[24];
    const float* p_vb   = (const float*)d_in[25];

    float* out = (float*)d_out;
    float* out_cond_p = out;                    // [32768]
    float* out_h      = out + 32768;            // [2048]
    float* out_c      = out + 34816;            // [2048]
    float* out_cov_g  = out + 36864;            // [32768]
    float* out_cov_p  = out + 69632;            // [32768]

    // workspace layout
    char* ws = (char*)d_ws;
    unsigned short* E16  = (unsigned short*)(ws);                        // 64 MB
    unsigned short* GW16 = (unsigned short*)(ws + 67108864ull);          //  2 MB
    unsigned short* PW16 = (unsigned short*)(ws + 69206016ull);          //  2 MB
    unsigned short* WH16 = (unsigned short*)(ws + 71303168ull);          // 64 MB
    float* tbuf  = (float*)(ws + 138412032ull);                          // 128 KB
    float* attn  = (float*)(ws + 138543104ull);                          // 128 KB
    float* zcomb = (float*)(ws + 138674176ull);                          // 4 KB
    float* ecomb = (float*)(ws + 138678272ull);                          // 4 KB
    float* ctx   = (float*)(ws + 138682368ull);                          // 4 KB
    float* mxsum = (float*)(ws + 138686464ull);                          // 8 B

    // 1) bf16 conversions
    k_cvt_bf16<<<4096, 256, 0, stream>>>(enc, E16, NROWS * DIM);
    k_cvt_bf16<<<1024, 256, 0, stream>>>(g_w1, GW16, DIM * DIM);
    k_cvt_bf16<<<1024, 256, 0, stream>>>(p_w1, PW16, DIM * DIM);

    // 2) LSTM: layer 0 then layer 1 (x of layer 1 = h of layer 0, read from d_out)
    k_lstm<<<1024, 256, 0, stream>>>(W_ih, W_hh, b_ih, b_hh,
                                     in_x, h0, c0, out_h, out_c);
    k_lstm<<<1024, 256, 0, stream>>>(W_ih + (size_t)4 * DIM * DIM, W_hh + (size_t)4 * DIM * DIM,
                                     b_ih + 4 * DIM, b_hh + 4 * DIM,
                                     out_h, h0 + DIM, c0 + DIM,
                                     out_h + DIM, out_c + DIM);
    const float* lstm_out = out_h + DIM;

    // 3) glimpse combined bias: z = g_w2@h + g_b2 ; zcomb = z + g_b1 + g_bc
    k_matvec<<<1024, 128, 0, stream>>>(g_w2, lstm_out, g_b2, g_b1, g_bc, zcomb);

    // 4) big GEMM #1: WH16 = E @ g_w1^T (bf16, fp32 accumulate)
    k_gemm_bf16<<<dim3(NROWS / 128, DIM / 64), 256, 0, stream>>>(E16, GW16, WH16);

    // 5) attention logits -> softmax -> coverage
    k_rowdot<<<NROWS, 256, 0, stream>>>(WH16, zcomb, g_wc, g_v, g_vb, cov_g, tbuf, 0);
    k_smax_reduce<<<1, 1024, 0, stream>>>(tbuf, mxsum);
    k_smax_apply<<<NROWS / 256, 256, 0, stream>>>(tbuf, mxsum, cov_g, attn, out_cov_g);

    // 6) context = attn^T @ w_h  (g_b1 folded in via init, sum(attn)==1)
    k_ctx_init<<<4, 256, 0, stream>>>(g_b1, ctx);
    k_ctx_accum<<<NROWS / 256, 256, 0, stream>>>(WH16, attn, ctx);

    // 7) pointer combined bias: e = p_w2@ctx + p_b2 ; ecomb = e + p_b1 + p_bc
    k_matvec<<<1024, 128, 0, stream>>>(p_w2, ctx, p_b2, p_b1, p_bc, ecomb);

    // 8) big GEMM #2: reuse WH16 buffer
    k_gemm_bf16<<<dim3(NROWS / 128, DIM / 64), 256, 0, stream>>>(E16, PW16, WH16);

    // 9) pointer logits (relu) -> softmax -> outputs
    k_rowdot<<<NROWS, 256, 0, stream>>>(WH16, ecomb, p_wc, p_v, p_vb, cov_p, tbuf, 1);
    k_smax_reduce<<<1, 1024, 0, stream>>>(tbuf, mxsum);
    k_smax_apply<<<NROWS / 256, 256, 0, stream>>>(tbuf, mxsum, cov_p, out_cond_p, out_cov_p);
}
